// AttentionBlock_49469433315393
// MI455X (gfx1250) — matrix operand order repaired and checked by reading, not compiled, not run on hardware
//
#include <hip/hip_runtime.h>
#include <hip/hip_bf16.h>

typedef __attribute__((ext_vector_type(16))) __bf16 v16bf;
typedef __attribute__((ext_vector_type(8)))  __bf16 v8bf;
typedef __attribute__((ext_vector_type(4)))  __bf16 v4bf;
typedef __attribute__((ext_vector_type(8)))  float  v8f;

#define B_   4
#define C_   512
#define HW_  4096
#define NH_  4
#define HD_  128

static __device__ inline v16bf cat8(v8bf a, v8bf b) {
  return __builtin_shufflevector(a, b, 0,1,2,3,4,5,6,7,8,9,10,11,12,13,14,15);
}
static __device__ inline v8f wmma_bf16(v16bf a, v16bf b, v8f c) {
  return __builtin_amdgcn_wmma_f32_16x16x32_bf16(false, a, false, b, (short)0, c, false, false);
}
static __device__ inline v8f v8f_zero() {
  v8f z = {0.f,0.f,0.f,0.f,0.f,0.f,0.f,0.f};
  return z;
}

// ---------------- Kernel 1: pack weights to bf16 (wq|wk|wv fused, wp) -------
__global__ __launch_bounds__(256) void k_convw(const float* wq, const float* wk,
                                               const float* wv, const float* wp,
                                               __bf16* wqkv, __bf16* wpb) {
  int i = blockIdx.x * 256 + threadIdx.x;          // 0 .. 1048575
  if (i < 1536 * 512) {
    int o = i >> 9;
    int c = i & 511;
    const float* src = (o < 512) ? wq : ((o < 1024) ? wk : wv);
    int oc = o & 511;
    wqkv[i] = (__bf16)src[oc * 512 + c];
  } else {
    int j = i - 1536 * 512;                        // 0 .. 262143
    wpb[j] = (__bf16)wp[j];
  }
}

// ---------------- Kernel 2: GroupNorm statistics ----------------------------
__global__ __launch_bounds__(256) void k_gn_stats(const float* x, float2* stats) {
  int bg = blockIdx.x;                              // b*32 + g, 0..127
  const float4* p = (const float4*)(x + (size_t)bg * 65536);
  float s = 0.f, sq = 0.f;
  for (int i = threadIdx.x; i < 16384; i += 256) {
    float4 v = p[i];
    s  += v.x + v.y + v.z + v.w;
    sq += v.x * v.x + v.y * v.y + v.z * v.z + v.w * v.w;
  }
  for (int off = 16; off; off >>= 1) { s += __shfl_xor(s, off); sq += __shfl_xor(sq, off); }
  __shared__ float rs[8], rq[8];
  int w = threadIdx.x >> 5;
  if ((threadIdx.x & 31) == 0) { rs[w] = s; rq[w] = sq; }
  __syncthreads();
  if (threadIdx.x == 0) {
    float S = 0.f, Q = 0.f;
    for (int i = 0; i < 8; ++i) { S += rs[i]; Q += rq[i]; }
    float mean = S * (1.f / 65536.f);
    float var  = Q * (1.f / 65536.f) - mean * mean;
    stats[bg] = make_float2(mean, rsqrtf(var + 1e-5f));
  }
}

// ---------------- Kernel 3: normalize + bf16 (channel-major) ----------------
__global__ __launch_bounds__(256) void k_gn_norm(const float* x, const float* gw,
                                                 const float* gb, const float2* stats,
                                                 __bf16* xn) {
  size_t vi = (size_t)blockIdx.x * 256 + threadIdx.x;   // 0 .. 2097151 (float4 units)
  size_t e  = vi * 4;
  int c  = (int)((e >> 12) & 511);
  int bg = (int)(e >> 16);
  float2 st = stats[bg];
  float g  = gw[c] * st.y;
  float bb = gb[c] - st.x * g;
  float4 v = *(const float4*)(x + e);
  v4bf o;
  o[0] = (__bf16)(v.x * g + bb);
  o[1] = (__bf16)(v.y * g + bb);
  o[2] = (__bf16)(v.z * g + bb);
  o[3] = (__bf16)(v.w * g + bb);
  *(v4bf*)(xn + e) = o;
}

// ---------------- Kernel 4: fused QKV GEMM (WMMA, double-buffered LDS) ------
// D[t, o] = sum_c xn[t,c] * Wqkv[o,c];  o in [0,1536)
__global__ __launch_bounds__(256) void k_qkv(const __bf16* __restrict__ xn,
                                             const __bf16* __restrict__ wqkv,
                                             const float* bq, const float* bk, const float* bv,
                                             __bf16* q, __bf16* k, __bf16* v) {
  __shared__ __bf16 sA[2][128][40];   // tokens x c-chunk
  __shared__ __bf16 sB[2][128][40];   // out-ch x c-chunk
  int tid = threadIdx.x;
  int bx = blockIdx.x;             // 0..127 token tiles (4 b * 32)
  int by = blockIdx.y;             // 0..11  o tiles
  int b  = bx >> 5;
  int t0 = (bx & 31) << 7;
  int o0 = by << 7;
  int lane = tid & 31, w = tid >> 5;
  int hf = lane >> 4, mn = lane & 15;
  int t_off = (w >> 2) * 64, o_off = (w & 3) * 32;
  const size_t xbase = (size_t)b * 512 * 4096;

  int a_cl = tid >> 4;             // 0..15 (c within chunk, +16 per pass)
  int a_tl = (tid & 15) * 8;       // token sub-offset
  int b_ol = tid >> 2;             // 0..63 (o row, +64 per pass)
  int b_c8 = (tid & 3) * 8;        // c sub-offset

  v8bf aR[2], bR[2];
  auto g_load = [&](int kk) {
#pragma unroll
    for (int p = 0; p < 2; ++p) {
      aR[p] = *(const v8bf*)(xn + xbase + (size_t)(kk + a_cl + p * 16) * 4096 + t0 + a_tl);
      bR[p] = *(const v8bf*)(wqkv + (size_t)(o0 + b_ol + p * 64) * 512 + kk + b_c8);
    }
  };
  auto s_store = [&](int buf) {
#pragma unroll
    for (int p = 0; p < 2; ++p) {
#pragma unroll
      for (int i = 0; i < 8; ++i) sA[buf][a_tl + i][a_cl + p * 16] = aR[p][i];
      *(v8bf*)&sB[buf][b_ol + p * 64][b_c8] = bR[p];
    }
  };

  v8f acc[4][2];
#pragma unroll
  for (int i = 0; i < 4; ++i)
#pragma unroll
    for (int j = 0; j < 2; ++j) acc[i][j] = v8f_zero();

  g_load(0);
  s_store(0);
  for (int it = 0; it < 16; ++it) {
    int cur = it & 1;
    if (it < 15) g_load((it + 1) * 32);
    __syncthreads();

    v16bf a[4], bo[2];
#pragma unroll
    for (int mt = 0; mt < 4; ++mt) {
      const __bf16* rp = &sA[cur][t_off + mt * 16 + mn][0];
      a[mt] = cat8(*(const v8bf*)(rp + 8 * hf), *(const v8bf*)(rp + 16 + 8 * hf));
    }
#pragma unroll
    for (int nt = 0; nt < 2; ++nt) {
      const __bf16* rp = &sB[cur][o_off + nt * 16 + mn][0];
      bo[nt] = cat8(*(const v8bf*)(rp + 8 * hf), *(const v8bf*)(rp + 16 + 8 * hf));
    }
#pragma unroll
    for (int mt = 0; mt < 4; ++mt)
#pragma unroll
      for (int nt = 0; nt < 2; ++nt)
        acc[mt][nt] = wmma_bf16(a[mt], bo[nt], acc[mt][nt]);

    if (it < 15) s_store(cur ^ 1);
  }

  // epilogue: + bias, convert bf16, scatter to [b,h,t,d]
#pragma unroll
  for (int nt = 0; nt < 2; ++nt) {
    int og   = o0 + o_off + nt * 16 + mn;
    int proj = og >> 9;
    int oc   = og & 511;
    int head = oc >> 7, d = oc & 127;
    const float* bias = (proj == 0) ? bq : ((proj == 1) ? bk : bv);
    __bf16* dst = (proj == 0) ? q : ((proj == 1) ? k : v);
    float bval = bias[oc];
    size_t dbase = (size_t)(b * 4 + head) * 4096 * 128 + d;
#pragma unroll
    for (int mt = 0; mt < 4; ++mt) {
      int trow = t0 + t_off + mt * 16 + 8 * hf;
#pragma unroll
      for (int r = 0; r < 8; ++r)
        dst[dbase + (size_t)(trow + r) * 128] = (__bf16)(acc[mt][nt][r] + bval);
    }
  }
}

// ---------------- Kernel 5: flash attention (WMMA, double-buffered K/V) -----
__global__ __launch_bounds__(256) void k_attn(const __bf16* __restrict__ q,
                                              const __bf16* __restrict__ k,
                                              const __bf16* __restrict__ v,
                                              __bf16* O) {
  __shared__ __bf16 sQ[128][136];      // q-rows x d
  __shared__ __bf16 sK[2][32][136];    // k-rows x d
  __shared__ __bf16 sVt[2][128][40];   // d x k-cols (transposed)
  __shared__ __bf16 sP[8][16][40];     // per-wave P staging (C->A layout)
  int tid = threadIdx.x;
  int bh  = blockIdx.y;             // 0..15
  int q0  = blockIdx.x << 7;        // q block base
  int lane = tid & 31, w = tid >> 5;
  int hf = lane >> 4, mn = lane & 15;
  size_t base = (size_t)bh * 4096 * 128;

  int rl = tid >> 4;                // 0..15 (+16 per pass)
  int d8 = (tid & 15) * 8;

  // load Q block
#pragma unroll
  for (int p = 0; p < 8; ++p) {
    int row = rl + p * 16;
    *(v8bf*)&sQ[row][d8] = *(const v8bf*)(q + base + (size_t)(q0 + row) * 128 + d8);
  }

  v8bf kR[2], vR[2];
  auto g_load = [&](int kb) {
#pragma unroll
    for (int p = 0; p < 2; ++p) {
      int row = rl + p * 16;
      kR[p] = *(const v8bf*)(k + base + (size_t)(kb + row) * 128 + d8);
      vR[p] = *(const v8bf*)(v + base + (size_t)(kb + row) * 128 + d8);
    }
  };
  auto s_store = [&](int buf) {
#pragma unroll
    for (int p = 0; p < 2; ++p) {
      int row = rl + p * 16;
      *(v8bf*)&sK[buf][row][d8] = kR[p];
#pragma unroll
      for (int i = 0; i < 8; ++i) sVt[buf][d8 + i][row] = vR[p][i];
    }
  };

  g_load(0);
  s_store(0);
  __syncthreads();

  v16bf qa[4];
#pragma unroll
  for (int kc = 0; kc < 4; ++kc) {
    const __bf16* rp = &sQ[w * 16 + mn][kc * 32];
    qa[kc] = cat8(*(const v8bf*)(rp + 8 * hf), *(const v8bf*)(rp + 16 + 8 * hf));
  }

  v8f o[8];
  float m[8], l[8];
#pragma unroll
  for (int i = 0; i < 8; ++i) { o[i] = v8f_zero(); m[i] = -INFINITY; l[i] = 0.f; }
  const float sc2 = 0.08838834764831845f * 1.4426950408889634f;  // 128^-0.5 * log2(e)

  for (int it = 0; it < 128; ++it) {
    int cur = it & 1;
    if (it < 127) g_load((it + 1) * 32);
    __syncthreads();

    // S = Q * K^T  (16q x 32k per wave)
    v8f s0 = v8f_zero(), s1 = v8f_zero();
#pragma unroll
    for (int kc = 0; kc < 4; ++kc) {
      const __bf16* r0 = &sK[cur][mn][kc * 32 + 8 * hf];
      const __bf16* r1 = &sK[cur][16 + mn][kc * 32 + 8 * hf];
      s0 = wmma_bf16(qa[kc], cat8(*(const v8bf*)r0, *(const v8bf*)(r0 + 16)), s0);
      s1 = wmma_bf16(qa[kc], cat8(*(const v8bf*)r1, *(const v8bf*)(r1 + 16)), s1);
    }

    // online softmax (row = r + 8*hf, 16-lane reductions)
#pragma unroll
    for (int r = 0; r < 8; ++r) {
      float mx = fmaxf(s0[r], s1[r]);
      mx = fmaxf(mx, __shfl_xor(mx, 1));
      mx = fmaxf(mx, __shfl_xor(mx, 2));
      mx = fmaxf(mx, __shfl_xor(mx, 4));
      mx = fmaxf(mx, __shfl_xor(mx, 8));
      float mnew = fmaxf(m[r], mx);
      float corr = exp2f((m[r] - mnew) * sc2);
      m[r] = mnew;
      float p0 = exp2f((s0[r] - mnew) * sc2);
      float p1 = exp2f((s1[r] - mnew) * sc2);
      sP[w][r + 8 * hf][mn]      = (__bf16)p0;
      sP[w][r + 8 * hf][16 + mn] = (__bf16)p1;
      float rs = p0 + p1;
      rs += __shfl_xor(rs, 1);
      rs += __shfl_xor(rs, 2);
      rs += __shfl_xor(rs, 4);
      rs += __shfl_xor(rs, 8);
      l[r] = l[r] * corr + rs;
#pragma unroll
      for (int dt = 0; dt < 8; ++dt) o[dt][r] *= corr;
    }

    // P (16x32) * V (32x128)
    const __bf16* pp = &sP[w][mn][0];
    v16bf pa = cat8(*(const v8bf*)(pp + 8 * hf), *(const v8bf*)(pp + 16 + 8 * hf));
#pragma unroll
    for (int dt = 0; dt < 8; ++dt) {
      const __bf16* vr = &sVt[cur][dt * 16 + mn][8 * hf];
      o[dt] = wmma_bf16(pa, cat8(*(const v8bf*)vr, *(const v8bf*)(vr + 16)), o[dt]);
    }

    if (it < 127) s_store(cur ^ 1);
  }

  // normalize and store token-major bf16 [b][t][512]
  int b = bh >> 2, head = bh & 3;
#pragma unroll
  for (int r = 0; r < 8; ++r) {
    int t = q0 + w * 16 + r + 8 * hf;
    float inv = 1.0f / l[r];
    size_t ob = ((size_t)b * 4096 + t) * 512 + head * 128 + mn;
#pragma unroll
    for (int dt = 0; dt < 8; ++dt) O[ob + dt * 16] = (__bf16)(o[dt][r] * inv);
  }
}

// ---------------- Kernel 6: output projection + residual (WMMA) -------------
// D[c, t] = sum_o Wp[c,o] * O[t,o];  out = x + D + bp
__global__ __launch_bounds__(256) void k_proj(const __bf16* __restrict__ O,
                                              const __bf16* __restrict__ wpb,
                                              const float* bp, const float* x, float* out) {
  __shared__ __bf16 sA[2][128][40];   // c x o-chunk
  __shared__ __bf16 sB[2][128][40];   // t x o-chunk
  int tid = threadIdx.x;
  int c0 = blockIdx.x << 7;        // 0..3
  int t0 = blockIdx.y << 7;        // 0..31
  int b  = blockIdx.z;             // 0..3
  int lane = tid & 31, w = tid >> 5;
  int hf = lane >> 4, mn = lane & 15;
  int c_off = (w >> 2) * 64, t_off = (w & 3) * 32;

  int rl = tid >> 2;               // 0..63 (+64 per pass)
  int o8 = (tid & 3) * 8;

  v8bf aR[2], bR[2];
  auto g_load = [&](int kk) {
#pragma unroll
    for (int p = 0; p < 2; ++p) {
      aR[p] = *(const v8bf*)(wpb + (size_t)(c0 + rl + p * 64) * 512 + kk + o8);
      bR[p] = *(const v8bf*)(O + ((size_t)b * 4096 + t0 + rl + p * 64) * 512 + kk + o8);
    }
  };
  auto s_store = [&](int buf) {
#pragma unroll
    for (int p = 0; p < 2; ++p) {
      *(v8bf*)&sA[buf][rl + p * 64][o8] = aR[p];
      *(v8bf*)&sB[buf][rl + p * 64][o8] = bR[p];
    }
  };

  v8f acc[4][2];
#pragma unroll
  for (int i = 0; i < 4; ++i)
#pragma unroll
    for (int j = 0; j < 2; ++j) acc[i][j] = v8f_zero();

  g_load(0);
  s_store(0);
  for (int it = 0; it < 16; ++it) {
    int cur = it & 1;
    if (it < 15) g_load((it + 1) * 32);
    __syncthreads();

    v16bf a[4], bo[2];
#pragma unroll
    for (int mt = 0; mt < 4; ++mt) {
      const __bf16* rp = &sA[cur][c_off + mt * 16 + mn][0];
      a[mt] = cat8(*(const v8bf*)(rp + 8 * hf), *(const v8bf*)(rp + 16 + 8 * hf));
    }
#pragma unroll
    for (int nt = 0; nt < 2; ++nt) {
      const __bf16* rp = &sB[cur][t_off + nt * 16 + mn][0];
      bo[nt] = cat8(*(const v8bf*)(rp + 8 * hf), *(const v8bf*)(rp + 16 + 8 * hf));
    }
#pragma unroll
    for (int mt = 0; mt < 4; ++mt)
#pragma unroll
      for (int nt = 0; nt < 2; ++nt)
        acc[mt][nt] = wmma_bf16(a[mt], bo[nt], acc[mt][nt]);

    if (it < 15) s_store(cur ^ 1);
  }

  // epilogue: residual add, coalesced along t
#pragma unroll
  for (int nt = 0; nt < 2; ++nt) {
    int tcol = t0 + t_off + nt * 16 + mn;
#pragma unroll
    for (int mt = 0; mt < 4; ++mt) {
#pragma unroll
      for (int r = 0; r < 8; ++r) {
        int c = c0 + c_off + mt * 16 + r + 8 * hf;
        size_t idx = ((size_t)b * 512 + c) * 4096 + tcol;
        out[idx] = x[idx] + acc[mt][nt][r] + bp[c];
      }
    }
  }
}

// ---------------- host launcher ---------------------------------------------
extern "C" void kernel_launch(void* const* d_in, const int* in_sizes, int n_in,
                              void* d_out, int out_size, void* d_ws, size_t ws_size,
                              hipStream_t stream) {
  const float* x    = (const float*)d_in[0];
  const float* gn_w = (const float*)d_in[1];
  const float* gn_b = (const float*)d_in[2];
  const float* wq   = (const float*)d_in[3];
  const float* bq   = (const float*)d_in[4];
  const float* wk   = (const float*)d_in[5];
  const float* bk   = (const float*)d_in[6];
  const float* wv   = (const float*)d_in[7];
  const float* bv   = (const float*)d_in[8];
  const float* wp   = (const float*)d_in[9];
  const float* bp   = (const float*)d_in[10];
  float* out = (float*)d_out;
  char* ws = (char*)d_ws;

  const size_t SZ_STATS = 1024;                       // 128 * float2
  const size_t SZ_WQKV  = (size_t)1536 * 512 * 2;
  const size_t SZ_WP    = (size_t)512 * 512 * 2;
  const size_t SZ_XN    = (size_t)B_ * C_ * HW_ * 2;  // 16 MB
  const size_t SZ_HVT   = (size_t)B_ * NH_ * HW_ * HD_ * 2;

  float2* stats = (float2*)ws;
  __bf16* wqkvb = (__bf16*)(ws + SZ_STATS);
  __bf16* wpb   = (__bf16*)(ws + SZ_STATS + SZ_WQKV);
  size_t off = SZ_STATS + SZ_WQKV + SZ_WP;
  __bf16* xn = (__bf16*)(ws + off); off += SZ_XN;
  __bf16* qb = (__bf16*)(ws + off); off += SZ_HVT;
  __bf16* kb = (__bf16*)(ws + off); off += SZ_HVT;
  __bf16* vb = (__bf16*)(ws + off); off += SZ_HVT;
  __bf16* Ob = (__bf16*)(ws + off);

  k_convw<<<4096, 256, 0, stream>>>(wq, wk, wv, wp, wqkvb, wpb);
  k_gn_stats<<<128, 256, 0, stream>>>(x, stats);
  k_gn_norm<<<8192, 256, 0, stream>>>(x, gn_w, gn_b, stats, xn);
  k_qkv<<<dim3(128, 12), 256, 0, stream>>>(xn, wqkvb, bq, bk, bv, qb, kb, vb);
  k_attn<<<dim3(32, 16), 256, 0, stream>>>(qb, kb, vb, Ob);
  k_proj<<<dim3(4, 32, 4), 256, 0, stream>>>(Ob, wpb, bp, x, out);
}
